// LineGrouper_87471303950752
// MI455X (gfx1250) — matrix operand order repaired and checked by reading, not compiled, not run on hardware
//
#include <hip/hip_runtime.h>
#include <stdint.h>

// Problem constants (fixed by the reference).
#define E_EDGES 262144
#define HID     256      // node feature width
#define N1      512      // layer-1 output width (2*HID)
#define NCLS    2
#define M_TILE  64       // edges per workgroup
#define THREADS 256      // 8 waves (wave32)
#define WAVES   8
#define W1P_U32 (32 * 8192)          // 32 blocks x 512 cols x 16 kpairs
#define BT_STRIDE 20                 // padded column stride (16 + 4 dwords)
#define BT_U32    (N1 * BT_STRIDE)   // 10240 dwords = 40 KB per buffer

typedef unsigned short u16;
typedef unsigned int   u32;
typedef __attribute__((ext_vector_type(16))) __bf16 v16bf;
typedef __attribute__((ext_vector_type(8)))  __bf16 v8bf;
typedef __attribute__((ext_vector_type(16))) u16    v16u;
typedef __attribute__((ext_vector_type(8)))  float  v8f;
typedef __attribute__((ext_vector_type(4)))  u32    v4u;
typedef __attribute__((ext_vector_type(8)))  int    v8i;
typedef __attribute__((ext_vector_type(4)))  int    v4i;

// f32 -> bf16 round-to-nearest-even (used only in the one-shot pack kernels)
static __device__ __forceinline__ u16 f2bf(float f) {
  u32 u = __float_as_uint(f);
  u32 r = u + 0x7FFFu + ((u >> 16) & 1u);
  return (u16)(r >> 16);
}
static __device__ __forceinline__ u32 pack2(float lo, float hi) {
  return (u32)f2bf(lo) | ((u32)f2bf(hi) << 16);
}
// packed-bf16 elementwise ops for the feature sections (v_pk_* path)
static __device__ __forceinline__ uint4 mul4(uint4 a, uint4 b) {
  v8bf r = __builtin_bit_cast(v8bf, a) * __builtin_bit_cast(v8bf, b);
  return __builtin_bit_cast(uint4, r);
}
static __device__ __forceinline__ uint4 absdiff4(uint4 a, uint4 b) {
  v8bf d = __builtin_bit_cast(v8bf, a) - __builtin_bit_cast(v8bf, b);
  uint4 r = __builtin_bit_cast(uint4, d);
  r.x &= 0x7fff7fffu; r.y &= 0x7fff7fffu;
  r.z &= 0x7fff7fffu; r.w &= 0x7fff7fffu;   // |x| = clear bf16 sign bits
  return r;
}
static __device__ __forceinline__ v16bf mk16(uint4 a, uint4 b) {
  union { uint4 q[2]; v16u v; } u;
  u.q[0] = a; u.q[1] = b;
  return __builtin_bit_cast(v16bf, u.v);
}

// ---- Tensor Data Mover: 32KB (8192 dword) global -> LDS copy with LDS padding.
// D# per CDNA5 ISA: 1-D tile, data_size=4B, type=2,
// pad_enable=1, pad_interval=3 (every 16 dwords), pad_amount=3 (4 dwords):
// LDS image becomes 512 columns x 20-dword stride -> 16B-aligned, conflict-free
// ds_load_b128 fragment loads (each 4-bank group serves exactly 2 lanes, the
// hardware floor for a full-wave b128).
static __device__ __forceinline__ void tdm_load_block(u32 lds_off, const void* gp) {
  unsigned long long ga = (unsigned long long)(uintptr_t)gp;
  v4u g0;
  g0[0] = 1u;                               // count=1 (valid user descriptor)
  g0[1] = lds_off;                          // lds_addr (bytes)
  g0[2] = (u32)ga;                          // global_addr[31:0]
  g0[3] = (u32)(ga >> 32) | (2u << 30);     // global_addr[56:32] | type=2
  v8i g1;
  g1[0] = (int)((2u << 16) | (1u << 20) | (3u << 22) | (3u << 25));
                                            // data_size=4B, pad_en, int=16dw, amt=4dw
  g1[1] = (int)(8192u << 16);               // tensor_dim0 = 8192 (low 16 bits)
  g1[2] = (int)(1u << 16);                  // tensor_dim0 hi=0 | tensor_dim1=1
  g1[3] = (int)(8192u << 16);               // tensor_dim1 hi=0 | tile_dim0=8192
  g1[4] = 1;                                // tile_dim1=1, tile_dim2=0
  g1[5] = 8192;                             // tensor_dim0_stride = 8192
  g1[6] = 0;
  g1[7] = 0;
  v4i z4 = {0, 0, 0, 0};
#if defined(__clang_major__) && (__clang_major__ >= 23)
  v8i z8 = {0, 0, 0, 0, 0, 0, 0, 0};
  __builtin_amdgcn_tensor_load_to_lds(g0, g1, z4, z4, z8, 0);
#else
  __builtin_amdgcn_tensor_load_to_lds(g0, g1, z4, z4, 0);
#endif
}

// ---- one-shot packing kernels (run once per launch; ~27 MB of traffic) ----
__global__ void pack_bf16_k(const float* __restrict__ in, uint4* __restrict__ outp,
                            int n4) {
  int i = blockIdx.x * THREADS + threadIdx.x;
  if (i >= n4) return;
  const float4* p = (const float4*)in + (size_t)i * 2;
  float4 a = p[0], b = p[1];
  uint4 r;
  r.x = pack2(a.x, a.y); r.y = pack2(a.z, a.w);
  r.z = pack2(b.x, b.y); r.w = pack2(b.z, b.w);
  outp[i] = r;
}
// W1 [1024,512] f32 -> 32 blocks of the LDS image BT[col][kpair] bf16x2
// (unpadded in global; the TDM inserts the padding on the way into LDS).
// Block id t = kk8*4 + s covers W1 rows [s*256 + kk8*32, +32).
__global__ void pack_w1_k(const float* __restrict__ W1, u32* __restrict__ W1P) {
  int o = blockIdx.x * THREADS + threadIdx.x;   // 0..262143
  int blk    = o >> 13;                         // 0..31
  int within = o & 8191;
  int c  = within >> 4;                         // column 0..511
  int rp = within & 15;                         // k-pair 0..15
  int kk8 = blk >> 2;
  int s   = blk & 3;
  int row = s * HID + kk8 * 32 + 2 * rp;
  float f0 = W1[(size_t)row * N1 + c];
  float f1 = W1[(size_t)(row + 1) * N1 + c];
  W1P[o] = pack2(f0, f1);
}

// ---- main fused kernel ----
__global__ __launch_bounds__(THREADS, 1)
void edge_mlp_wmma(const u16*   __restrict__ nodebf,   // [50000][256] bf16
                   const int*   __restrict__ src,
                   const int*   __restrict__ dst,
                   const u32*   __restrict__ W1P,      // 32 x 32KB packed blocks
                   const float* __restrict__ b1g,
                   const float* __restrict__ W2,
                   const float* __restrict__ b2g,
                   float*       __restrict__ out) {
  __shared__ u32   BT[2][BT_U32];                  // 2 x 40KB, TDM double buffer
  __shared__ u16   Ai[M_TILE * 32];                // gathered h_i chunk (bf16)
  __shared__ u16   Aj[M_TILE * 32];                // gathered h_j chunk (bf16)
  __shared__ float partial[WAVES][M_TILE][NCLS];

  const int tid  = threadIdx.x;
  const int wave = tid >> 5;
  const int lane = tid & 31;
  const int l16  = lane & 15;
  const int hi   = lane >> 4;
  const int e0   = blockIdx.x * M_TILE;

  v8f acc[4][4];
  #pragma unroll
  for (int mt = 0; mt < 4; ++mt)
    #pragma unroll
    for (int nt = 0; nt < 4; ++nt)
      #pragma unroll
      for (int v = 0; v < 8; ++v)
        acc[mt][nt][v] = 0.f;

  // Fixed gather role per thread: row ar (edge), 8-element column group ac.
  const int ar = tid >> 2;
  const int ac = (tid & 3) * 8;
  const u16* nip = nodebf + (size_t)src[e0 + ar] * HID;
  const u16* njp = nodebf + (size_t)dst[e0 + ar] * HID;

  // Prologue: TDM copy of weight block 0 into buffer 0 (wave 0 only).
  if (wave == 0)
    tdm_load_block((u32)(uintptr_t)&BT[0][0], W1P);

  for (int kk8 = 0; kk8 < 8; ++kk8) {
    uint4 xi[4][2], xj[4][2];        // raw A fragments, live across 4 sections
    #pragma unroll
    for (int s = 0; s < 4; ++s) {    // feature section, compile-time
      const int t   = kk8 * 4 + s;
      const int buf = t & 1;
      __syncthreads();               // readers of BT[buf^1] / Ai / Aj are done
      if (s == 0) {                  // new 32-wide chunk of gathered features
        const int kk = kk8 * 32;
        *(uint4*)&Ai[ar * 32 + ac] = *(const uint4*)(nip + kk + ac); // 16B gather
        *(uint4*)&Aj[ar * 32 + ac] = *(const uint4*)(njp + kk + ac);
      }
      if (wave == 0) {
        if (t < 31) {                // async-prefetch next weight block
          tdm_load_block((u32)(uintptr_t)&BT[buf ^ 1][0],
                         W1P + (size_t)(t + 1) * 8192);
          __builtin_amdgcn_s_wait_tensorcnt(1);   // block t has landed
        } else {
          __builtin_amdgcn_s_wait_tensorcnt(0);
        }
      }
      __syncthreads();               // BT[buf] + Ai/Aj visible to all waves

      if (s == 0) {                  // hoist raw A fragments into registers
        #pragma unroll
        for (int mt = 0; mt < 4; ++mt) {
          const int r = mt * 16 + l16;
          xi[mt][0] = *(const uint4*)&Ai[r * 32 + hi * 16];
          xi[mt][1] = *(const uint4*)&Ai[r * 32 + 8 + hi * 16];
          xj[mt][0] = *(const uint4*)&Aj[r * 32 + hi * 16];
          xj[mt][1] = *(const uint4*)&Aj[r * 32 + 8 + hi * 16];
        }
      }
      // B fragments: lane = column n, 20-dword padded stride (aligned b128)
      v16bf bfrag[4];
      #pragma unroll
      for (int nt = 0; nt < 4; ++nt) {
        const int n = wave * 64 + nt * 16 + l16;
        uint4 blo = *(const uint4*)&BT[buf][n * BT_STRIDE + hi * 8];
        uint4 bhi = *(const uint4*)&BT[buf][n * BT_STRIDE + hi * 8 + 4];
        bfrag[nt] = mk16(blo, bhi);
      }
      // A variant (branch-free, s is compile-time) + 16 WMMAs
      #pragma unroll
      for (int mt = 0; mt < 4; ++mt) {
        v16bf afrag;
        if (s == 0)      afrag = mk16(xi[mt][0], xi[mt][1]);
        else if (s == 1) afrag = mk16(xj[mt][0], xj[mt][1]);
        else if (s == 2) afrag = mk16(absdiff4(xi[mt][0], xj[mt][0]),
                                      absdiff4(xi[mt][1], xj[mt][1]));
        else             afrag = mk16(mul4(xi[mt][0], xj[mt][0]),
                                      mul4(xi[mt][1], xj[mt][1]));
        #pragma unroll
        for (int nt = 0; nt < 4; ++nt)
          acc[mt][nt] = __builtin_amdgcn_wmma_f32_16x16x32_bf16(
              false, afrag, false, bfrag[nt],
              (short)0, acc[mt][nt], false, false);
      }
    }
  }

  // ---- epilogue: bias + ReLU, tiny GEMM vs W2 [512,2], cross-lane/wave reduce ----
  float w2c0[4], w2c1[4], b1n[4];
  #pragma unroll
  for (int nt = 0; nt < 4; ++nt) {
    const int n = wave * 64 + nt * 16 + l16;
    w2c0[nt] = W2[n * 2 + 0];
    w2c1[nt] = W2[n * 2 + 1];
    b1n[nt]  = b1g[n];
  }
  #pragma unroll
  for (int mt = 0; mt < 4; ++mt) {
    float p0[8], p1[8];
    #pragma unroll
    for (int v = 0; v < 8; ++v) { p0[v] = 0.f; p1[v] = 0.f; }
    #pragma unroll
    for (int nt = 0; nt < 4; ++nt)
      #pragma unroll
      for (int v = 0; v < 8; ++v) {
        float h = fmaxf(acc[mt][nt][v] + b1n[nt], 0.f);
        p0[v] += h * w2c0[nt];
        p1[v] += h * w2c1[nt];
      }
    #pragma unroll
    for (int m = 1; m < 16; m <<= 1)
      #pragma unroll
      for (int v = 0; v < 8; ++v) {
        p0[v] += __shfl_xor(p0[v], m, 32);
        p1[v] += __shfl_xor(p1[v], m, 32);
      }
    if (l16 == 0) {
      #pragma unroll
      for (int v = 0; v < 8; ++v) {
        partial[wave][mt * 16 + hi * 8 + v][0] = p0[v];
        partial[wave][mt * 16 + hi * 8 + v][1] = p1[v];
      }
    }
  }
  __syncthreads();
  if (tid < M_TILE * NCLS) {
    const int r = tid >> 1, c = tid & 1;
    float sum = b2g[c];
    #pragma unroll
    for (int w = 0; w < WAVES; ++w) sum += partial[w][r][c];
    out[(size_t)(e0 + r) * NCLS + c] = sum;
  }
}

extern "C" void kernel_launch(void* const* d_in, const int* in_sizes, int n_in,
                              void* d_out, int out_size, void* d_ws, size_t ws_size,
                              hipStream_t stream) {
  (void)in_sizes; (void)n_in; (void)out_size; (void)ws_size;
  const float* node = (const float*)d_in[0];
  const int*   src  = (const int*)d_in[1];
  const int*   dst  = (const int*)d_in[2];
  const float* W1   = (const float*)d_in[3];
  const float* b1   = (const float*)d_in[4];
  const float* W2   = (const float*)d_in[5];
  const float* b2   = (const float*)d_in[6];
  float* out = (float*)d_out;

  // Workspace layout: [0, 1MB) packed W1 blocks; [1MB, +25.6MB) bf16 node table.
  u32* W1P    = (u32*)d_ws;
  u16* nodebf = (u16*)((char*)d_ws + (size_t)W1P_U32 * sizeof(u32));

  pack_w1_k<<<dim3(W1P_U32 / THREADS), dim3(THREADS), 0, stream>>>(W1, W1P);
  const int n4 = 50000 * HID / 8;
  pack_bf16_k<<<dim3((n4 + THREADS - 1) / THREADS), dim3(THREADS), 0, stream>>>(
      node, (uint4*)nodebf, n4);
  edge_mlp_wmma<<<dim3(E_EDGES / M_TILE), dim3(THREADS), 0, stream>>>(
      nodebf, src, dst, W1P, b1, W2, b2, out);
}
